// MultiheadMaskedAttention_48816598286522
// MI455X (gfx1250) — compile-verified, matrix operand order checked by reading, not run on hardware
//
#include <hip/hip_runtime.h>
#include <hip/hip_bf16.h>

typedef __attribute__((ext_vector_type(16))) _Float16 v16h;
typedef __attribute__((ext_vector_type(8)))  _Float16 v8h;
typedef __attribute__((ext_vector_type(8)))  float    v8f;

#define NH    16
#define HD    64
#define SEQ   2048
#define BATCH 4
#define HID   1024
#define NTOK  (BATCH * SEQ)   // 8192

// ---------------------------------------------------------------------------
// WMMA helpers (CDNA5 wave32, 16x16x32 f16 -> f32)
// ---------------------------------------------------------------------------
__device__ __forceinline__ v8f wmma_f16(v16h a, v16h b, v8f c) {
  // (neg_a, A, neg_b, B, c_mod, C, reuse_a, reuse_b)
  return __builtin_amdgcn_wmma_f32_16x16x32_f16(false, a, false, b, (short)0, c,
                                                false, false);
}

// Load a 16-row x 32-half operand tile from row-major f16 memory into the
// CDNA5 WMMA A/B register layout:
//   lane r (0-15):  halfs[0..7]=row r K=0..7,   halfs[8..15]=row r K=16..23
//   lane r+16:      halfs[0..7]=row r K=8..15,  halfs[8..15]=row r K=24..31
// => two global_load_b128 per lane.
__device__ __forceinline__ v16h load_tile16x32(const _Float16* __restrict__ base,
                                               int row_stride /* halfs */) {
  const int lane = threadIdx.x & 31;
  const int r    = lane & 15;
  const int hi   = lane >> 4;
  const _Float16* p = base + (size_t)r * row_stride + hi * 8;
  v8h lo = *reinterpret_cast<const v8h*>(p);
  v8h hh = *reinterpret_cast<const v8h*>(p + 16);
  v16h out;
#pragma unroll
  for (int i = 0; i < 8; ++i) { out[i] = lo[i]; out[8 + i] = hh[i]; }
  return out;
}

__device__ __forceinline__ float lane_bcast(float v, int srcLane) {
  return __int_as_float(
      __builtin_amdgcn_ds_bpermute(srcLane << 2, __float_as_int(v)));
}

// ---------------------------------------------------------------------------
// f32 -> f16 conversion / transpose staging kernels (bandwidth-trivial)
// ---------------------------------------------------------------------------
__global__ void cvt_f16_kernel(const float* __restrict__ in,
                               _Float16* __restrict__ out, int n) {
  int i = blockIdx.x * blockDim.x + threadIdx.x;
  if (i < n) out[i] = (_Float16)in[i];
}

// out[n][k] = (f16) in[k][n]   (in: rows x cols, out: cols x rows)
__global__ void transpose_cvt_kernel(const float* __restrict__ in,
                                     _Float16* __restrict__ out,
                                     int rows, int cols) {
  int i = blockIdx.x * blockDim.x + threadIdx.x;
  if (i < rows * cols) {
    int k = i / cols, n = i % cols;
    out[(size_t)n * rows + k] = (_Float16)in[i];
  }
}

// ---------------------------------------------------------------------------
// QKV projection: (8192,1024) @ (1024,3072) + bias, scattering results into
//   Qh, Kh: (B, NH, S, HD) f16   (rows = tokens, perfect for WMMA tile loads)
//   Vt:     (B, NH, HD, S) f16   (transposed so P@V B-operand loads are clean)
// Each wave computes a 64x64 output tile (16 WMMAs per 32-deep K step).
// ---------------------------------------------------------------------------
__global__ void __launch_bounds__(256)
qkv_gemm_kernel(const _Float16* __restrict__ xh,    // (8192,1024)
                const _Float16* __restrict__ wt,    // (3072,1024) = W_qkv^T
                const float* __restrict__ bqkv,     // (3072)
                _Float16* __restrict__ qh,
                _Float16* __restrict__ kh,
                _Float16* __restrict__ vt) {
  const int wave = blockIdx.x * (blockDim.x >> 5) + (threadIdx.x >> 5);
  const int NT   = (3 * HID) / 64;            // 48 n-tiles
  const int m0   = (wave / NT) * 64;
  const int n0   = (wave % NT) * 64;
  const int lane = threadIdx.x & 31;

  v8f zero = {};
  v8f acc[4][4];
#pragma unroll
  for (int i = 0; i < 4; ++i)
#pragma unroll
    for (int j = 0; j < 4; ++j) acc[i][j] = zero;

  for (int k0 = 0; k0 < HID; k0 += 32) {
    v16h a[4], b[4];
#pragma unroll
    for (int i = 0; i < 4; ++i)
      a[i] = load_tile16x32(xh + (size_t)(m0 + i * 16) * HID + k0, HID);
#pragma unroll
    for (int j = 0; j < 4; ++j)
      b[j] = load_tile16x32(wt + (size_t)(n0 + j * 16) * HID + k0, HID);
#pragma unroll
    for (int i = 0; i < 4; ++i)
#pragma unroll
      for (int j = 0; j < 4; ++j) acc[i][j] = wmma_f16(a[i], b[j], acc[i][j]);
  }

  // Epilogue: D layout is lane n = lane&15 (col), VGPR v -> row (lane>>4)*8+v
  const int nlane = lane & 15;
  const int mbase = (lane >> 4) * 8;
#pragma unroll
  for (int j = 0; j < 4; ++j) {
    const int nb     = n0 + j * 16;        // column base in 0..3071
    const int region = nb / HID;           // 0=Q 1=K 2=V
    const int cin    = nb % HID;
    const int h      = cin / HD;
    const int d      = (cin % HD) + nlane;
    const float bias = bqkv[nb + nlane];
#pragma unroll
    for (int i = 0; i < 4; ++i) {
#pragma unroll
      for (int v = 0; v < 8; ++v) {
        const int t  = m0 + i * 16 + mbase + v;   // token 0..8191
        const int bb = t >> 11;
        const int s  = t & (SEQ - 1);
        const int bh = bb * NH + h;
        const float val = acc[i][j][v] + bias;
        if (region == 0)
          qh[((size_t)bh * SEQ + s) * HD + d] = (_Float16)val;
        else if (region == 1)
          kh[((size_t)bh * SEQ + s) * HD + d] = (_Float16)val;
        else
          vt[((size_t)bh * HD + d) * SEQ + s] = (_Float16)val;
      }
    }
  }
}

// ---------------------------------------------------------------------------
// Causal flash attention. One wave owns a 16-query tile of one (b,h).
// Scores computed transposed (Sᵀ = K_tile x Qᵀ): D[m=key][n=query], so the
// f16 probability pair for a 32-key block lands exactly in the A-operand
// layout of the P@V WMMA with zero shuffles.
// ---------------------------------------------------------------------------
__global__ void __launch_bounds__(256)
flash_attn_kernel(const _Float16* __restrict__ qh,
                  const _Float16* __restrict__ kh,
                  const _Float16* __restrict__ vt,
                  _Float16* __restrict__ zh) {   // (8192, 1024)
  const int wave = blockIdx.x * (blockDim.x >> 5) + (threadIdx.x >> 5);
  const int qt   = wave & 127;   // 128 query tiles per (b,h)
  const int bh   = wave >> 7;    // 0..63
  const int b    = bh >> 4;
  const int h    = bh & 15;
  const int q0   = qt * 16;
  const int lane = threadIdx.x & 31;
  const int hi   = lane >> 4;
  const int mbase = hi * 8;

  const _Float16* Qb = qh + ((size_t)bh * SEQ + q0) * HD;
  const v16h qa0 = load_tile16x32(Qb, HD);        // hd 0..31
  const v16h qa1 = load_tile16x32(Qb + 32, HD);   // hd 32..63

  v8f zero = {};
  v8f z[4] = {zero, zero, zero, zero};
  float m_i = -1e30f;
  float l_i = 0.0f;
  const float scale = 0.125f;                     // 1/sqrt(64)
  const int   qlane = q0 + (lane & 15);           // this lane's query row

  for (int kb = 0; kb <= q0 + 15; kb += 32) {
    const _Float16* Kb0 = kh + ((size_t)bh * SEQ + kb) * HD;
    v16h k0a = load_tile16x32(Kb0, HD);                  // keys kb..+15, hd lo
    v16h k0b = load_tile16x32(Kb0 + 32, HD);             //               hd hi
    v16h k1a = load_tile16x32(Kb0 + 16 * HD, HD);        // keys kb+16..+31
    v16h k1b = load_tile16x32(Kb0 + 16 * HD + 32, HD);

    v8f st0 = wmma_f16(k0a, qa0, zero);
    st0     = wmma_f16(k0b, qa1, st0);
    v8f st1 = wmma_f16(k1a, qa0, zero);
    st1     = wmma_f16(k1b, qa1, st1);

    // scale + causal mask (element: key = kb[+16] + mbase + v, query = qlane)
    if (kb + 31 > q0) {
#pragma unroll
      for (int v = 0; v < 8; ++v) {
        const int kk0 = kb + mbase + v;
        st0[v] = (kk0 > qlane)      ? -1e30f : st0[v] * scale;
        st1[v] = (kk0 + 16 > qlane) ? -1e30f : st1[v] * scale;
      }
    } else {
#pragma unroll
      for (int v = 0; v < 8; ++v) { st0[v] *= scale; st1[v] *= scale; }
    }

    // per-query running max (lane + lane^16 jointly hold the 32 keys)
    float mx = -1e30f;
#pragma unroll
    for (int v = 0; v < 8; ++v) mx = fmaxf(mx, fmaxf(st0[v], st1[v]));
    mx = fmaxf(mx, lane_bcast(mx, lane ^ 16));
    const float m_new = fmaxf(m_i, mx);
    const float corr  = __expf(m_i - m_new);

    // probabilities -> f16 A-operand for P@V, plus row sum
    float rs = 0.0f;
    v16h pa;
#pragma unroll
    for (int v = 0; v < 8; ++v) {
      const float p0 = __expf(st0[v] - m_new);
      const float p1 = __expf(st1[v] - m_new);
      rs += p0 + p1;
      pa[v]     = (_Float16)p0;
      pa[8 + v] = (_Float16)p1;
    }
    rs += lane_bcast(rs, lane ^ 16);
    l_i = l_i * corr + rs;
    m_i = m_new;

    // broadcast correction into accumulator row layout (row = mbase+v)
    float corrE[8];
#pragma unroll
    for (int v = 0; v < 8; ++v) corrE[v] = lane_bcast(corr, mbase + v);

    const _Float16* Vb = vt + (size_t)bh * HD * SEQ + kb;
#pragma unroll
    for (int t = 0; t < 4; ++t) {
      v16h vbt = load_tile16x32(Vb + (size_t)(t * 16) * SEQ, SEQ);
      v8f c;
#pragma unroll
      for (int v = 0; v < 8; ++v) c[v] = z[t][v] * corrE[v];
      z[t] = wmma_f16(pa, vbt, c);
    }
  }

  // normalize and store (D layout: col = lane&15, row = mbase+v)
  const float linv = 1.0f / l_i;
  float linvE[8];
#pragma unroll
  for (int v = 0; v < 8; ++v) linvE[v] = lane_bcast(linv, mbase + v);

  const int nlane = lane & 15;
#pragma unroll
  for (int t = 0; t < 4; ++t) {
#pragma unroll
    for (int v = 0; v < 8; ++v) {
      const int row = b * SEQ + q0 + mbase + v;
      const int col = h * HD + t * 16 + nlane;
      zh[(size_t)row * HID + col] = (_Float16)(z[t][v] * linvE[v]);
    }
  }
}

// ---------------------------------------------------------------------------
// Output projection: (8192,1024) @ (1024,1024) + b_o -> f32 out
// ---------------------------------------------------------------------------
__global__ void __launch_bounds__(256)
out_gemm_kernel(const _Float16* __restrict__ zh,   // (8192,1024)
                const _Float16* __restrict__ wot,  // (1024,1024) = W_o^T
                const float* __restrict__ bo,
                float* __restrict__ out) {
  const int wave = blockIdx.x * (blockDim.x >> 5) + (threadIdx.x >> 5);
  const int NT   = HID / 64;                 // 16 n-tiles
  const int m0   = (wave / NT) * 64;
  const int n0   = (wave % NT) * 64;
  const int lane = threadIdx.x & 31;

  v8f zero = {};
  v8f acc[4][4];
#pragma unroll
  for (int i = 0; i < 4; ++i)
#pragma unroll
    for (int j = 0; j < 4; ++j) acc[i][j] = zero;

  for (int k0 = 0; k0 < HID; k0 += 32) {
    v16h a[4], b[4];
#pragma unroll
    for (int i = 0; i < 4; ++i)
      a[i] = load_tile16x32(zh + (size_t)(m0 + i * 16) * HID + k0, HID);
#pragma unroll
    for (int j = 0; j < 4; ++j)
      b[j] = load_tile16x32(wot + (size_t)(n0 + j * 16) * HID + k0, HID);
#pragma unroll
    for (int i = 0; i < 4; ++i)
#pragma unroll
      for (int j = 0; j < 4; ++j) acc[i][j] = wmma_f16(a[i], b[j], acc[i][j]);
  }

  const int nlane = lane & 15;
  const int mbase = (lane >> 4) * 8;
#pragma unroll
  for (int j = 0; j < 4; ++j) {
    const int nc   = n0 + j * 16 + nlane;
    const float bb = bo[nc];
#pragma unroll
    for (int i = 0; i < 4; ++i) {
#pragma unroll
      for (int v = 0; v < 8; ++v) {
        const int m = m0 + i * 16 + mbase + v;
        out[(size_t)m * HID + nc] = acc[i][j][v] + bb;
      }
    }
  }
}

// ---------------------------------------------------------------------------
// Launch
// ---------------------------------------------------------------------------
extern "C" void kernel_launch(void* const* d_in, const int* in_sizes, int n_in,
                              void* d_out, int out_size, void* d_ws,
                              size_t ws_size, hipStream_t stream) {
  const float* x    = (const float*)d_in[0];
  const float* wqkv = (const float*)d_in[1];
  const float* bqkv = (const float*)d_in[2];
  const float* wo   = (const float*)d_in[3];
  const float* bo   = (const float*)d_in[4];
  float* out        = (float*)d_out;

  char* ws = (char*)d_ws;
  size_t off = 0;
  auto carve = [&](size_t bytes) {
    void* p = ws + off;
    off += (bytes + 255) & ~(size_t)255;
    return p;
  };

  _Float16* xh    = (_Float16*)carve((size_t)NTOK * HID * 2);        // 16 MB
  _Float16* wqkvt = (_Float16*)carve((size_t)3 * HID * HID * 2);     //  6 MB
  _Float16* wot   = (_Float16*)carve((size_t)HID * HID * 2);         //  2 MB
  _Float16* qh    = (_Float16*)carve((size_t)NTOK * HID * 2);        // 16 MB
  _Float16* kh    = (_Float16*)carve((size_t)NTOK * HID * 2);        // 16 MB
  _Float16* vt    = (_Float16*)carve((size_t)NTOK * HID * 2);        // 16 MB
  _Float16* zh    = (_Float16*)carve((size_t)NTOK * HID * 2);        // 16 MB

  // Stage: f32 -> f16 (+ weight transposes)
  {
    int n = NTOK * HID;
    cvt_f16_kernel<<<(n + 255) / 256, 256, 0, stream>>>(x, xh, n);
    int nq = HID * 3 * HID;
    transpose_cvt_kernel<<<(nq + 255) / 256, 256, 0, stream>>>(wqkv, wqkvt, HID,
                                                               3 * HID);
    int no = HID * HID;
    transpose_cvt_kernel<<<(no + 255) / 256, 256, 0, stream>>>(wo, wot, HID,
                                                               HID);
  }

  // QKV projection: (8192/64)*(3072/64) = 6144 waves, 8 waves/block
  qkv_gemm_kernel<<<6144 / 8, 256, 0, stream>>>(xh, wqkvt, bqkv, qh, kh, vt);

  // Flash attention: 4*16*128 = 8192 wave-tiles, 8 waves/block
  flash_attn_kernel<<<8192 / 8, 256, 0, stream>>>(qh, kh, vt, zh);

  // Output projection: (8192/64)*(1024/64) = 2048 waves, 8 waves/block
  out_gemm_kernel<<<2048 / 8, 256, 0, stream>>>(zh, wot, bo, out);
}
